// SynchronizationLoss_79431125172891
// MI455X (gfx1250) — matrix operand order repaired and checked by reading, not compiled, not run on hardware
//
#include <hip/hip_runtime.h>
#include <stdint.h>

typedef __attribute__((ext_vector_type(16))) _Float16 v16h;
typedef __attribute__((ext_vector_type(8)))  float    v8f;
typedef __attribute__((ext_vector_type(8)))  uint32_t v8u;
typedef __attribute__((ext_vector_type(4)))  float    v4f;
typedef __attribute__((ext_vector_type(4)))  _Float16 v4h;
typedef __attribute__((ext_vector_type(4)))  uint32_t v4u;

#define N_TRIALS 8
#define T_TOTAL  600
#define N_NEUR   30000
#define T_WIN    500
#define NS       50
#define NSP      64            // sample columns padded to 64 (4 WMMA N-tiles)
#define MAXC     200
#define KC       256           // K-chunk per LDS stage (8 WMMA steps)
#define NCHUNK   118           // ceil(30000/256); last chunk K-guarded
#define NSEG     6             // K-split factor for occupancy (exact-int sums => order-invariant)
#define NPART    (N_TRIALS*NSEG)
#define AST      (KC+8)        // LDS row stride in halves: 528 B = 132 words (== 4 mod 64 banks)
#define BST      (KC+8)
#define NBINS    16

// BIN_MS = round(logspace(-3,0,20)*1000) for b < 0.25 ; nb = 500 // bs
__constant__ int c_bin[NBINS] = {1,1,2,3,4,6,9,13,18,26,38,55,78,113,162,234};
__constant__ int c_nb[NBINS]  = {500,500,250,166,125,83,55,38,27,19,13,9,6,4,3,2};

// ---------------- Stage 0: zero the f16 weight buffer ----------------
__global__ void zero_ws_kernel(uint32_t* p, int n) {
  for (int i = blockIdx.x * blockDim.x + threadIdx.x; i < n; i += gridDim.x * blockDim.x)
    p[i] = 0u;
}

// ---------------- Stage 1: scatter sample ids into dense FP16 weights ----------
// W[trial r][sample s (pad 64)][neuron k]; entries are small ints (<=200), exact in f16.
// One thread per sample => private column => no atomics, deterministic.
__global__ void build_w_kernel(const int* __restrict__ trials,
                               const long long* __restrict__ ids,
                               const float* __restrict__ mask,
                               _Float16* __restrict__ W) {
  int s = threadIdx.x;
  if (s >= NS) return;
  int r = trials[s];
  _Float16* row = W + ((size_t)r * NSP + s) * (size_t)N_NEUR;
  for (int j = 0; j < MAXC; ++j) {
    float m = mask[s * MAXC + j];
    if (m != 0.0f) {
      long long id = ids[s * MAXC + j];
      row[id] = (_Float16)((float)row[id] + m);   // ids may repeat; serial RMW is safe
    }
  }
}

// ---------------- Stage 2: partial[r,seg][t][s] = sp[r,t,Kseg] . W_r[Kseg,s] ----------
// Block = 256 threads = 8 wave32s; block owns a 32(M) x 64(N) tile, wave (mt,nt) a 16x16
// sub-tile. grid = (16 Mblocks, 8 trials, 6 K-segments). A: vectorized NT b128 loads
// (f32->f16). B: async DMA global->LDS (f16, no conversion). All sums are exact integers.
__global__ __launch_bounds__(256) void gemm_sel_kernel(const float* __restrict__ spikes,
                                                       const _Float16* __restrict__ W,
                                                       float* __restrict__ part) {
  __shared__ __align__(16) _Float16 As[32  * AST];   // ~16.5 KB
  __shared__ __align__(16) _Float16 Bs[NSP * BST];   // ~33 KB

  const int tid  = threadIdx.x;
  const int lane = tid & 31;
  const int wv   = tid >> 5;
  const int mt   = wv >> 2, nt = wv & 3;
  const int lg   = lane >> 4, ml = lane & 15;
  const int m0   = blockIdx.x * 32;
  const int r    = blockIdx.y;
  const int seg  = blockIdx.z;

  const float*    spr = spikes + (size_t)r * T_TOTAL * N_NEUR;  // sp[r,t,k], k contiguous
  const _Float16* wr  = W      + (size_t)r * NSP * N_NEUR;      // W_r[s][k], k contiguous

  v8f acc = {};

  for (int c = seg; c < NCHUNK; c += NSEG) {
    const int k0 = c * KC;
    __syncthreads();   // previous chunk fully consumed

    // ---- A tile: 32 rows x 256 cols, 8 x b128 NT loads/thread, f32->f16, b64 DS stores.
    // 30000 % 4 == 0 => a float4 never straddles the K edge; guard is per-vector.
    #pragma unroll
    for (int i = 0; i < 8; ++i) {
      int q  = i * 256 + tid;
      int m  = q >> 6;                 // 64 float4 per row
      int kc = (q & 63) << 2;
      int t  = m0 + m, k = k0 + kc;
      v4f v = {0.f, 0.f, 0.f, 0.f};
      if (t < T_WIN && k < N_NEUR)
        v = __builtin_nontemporal_load((const v4f*)(spr + (size_t)t * N_NEUR + k));
      v4h h;
      h.x = (_Float16)v.x; h.y = (_Float16)v.y; h.z = (_Float16)v.z; h.w = (_Float16)v.w;
      *(v4h*)&As[m * AST + kc] = h;
    }

    // ---- B tile: 64 rows x 256 cols f16, 8 x b128 async DMA global->LDS per thread.
    // 30000 % 8 == 0 => an 8-half vector never straddles the K edge.
    #pragma unroll
    for (int i = 0; i < 8; ++i) {
      int q  = i * 256 + tid;
      int s  = q >> 5;                 // 32 x 16B per row
      int kc = (q & 31) << 3;
      int k  = k0 + kc;
      if (k < N_NEUR) {
        uint32_t ldsa = (uint32_t)(uintptr_t)&Bs[s * BST + kc];     // low 32 bits = LDS offset
        uint32_t voff = (uint32_t)(((size_t)s * N_NEUR + k) * 2u);  // byte offset from wr
        asm volatile("global_load_async_to_lds_b128 %0, %1, %2"
                     :: "v"(ldsa), "v"(voff), "s"(wr) : "memory");
      } else {
        v4u z = {0u, 0u, 0u, 0u};
        *(v4u*)&Bs[s * BST + kc] = z;
      }
    }
    asm volatile("s_wait_asynccnt 0x0" ::: "memory");
    __syncthreads();

    // ---- 8 WMMA steps over the staged K chunk
    #pragma unroll
    for (int st = 0; st < KC / 32; ++st) {
      const int kb = st * 32;
      const _Float16* arow = &As[(mt * 16 + ml) * AST + kb];
      const _Float16* brow = &Bs[(nt * 16 + ml) * BST + kb];
      v8u au, bu;
      #pragma unroll
      for (int g = 0; g < 8; ++g) {
        // A 16x32 f16 layout (ISA 7.12.2): K = (g>=4?16:0) + laneHalf*8 + (g&3)*2 (pair)
        int ka  = ((g & 4) << 2) + (lg << 3) + ((g & 3) << 1);
        au[g] = *reinterpret_cast<const uint32_t*>(arow + ka);
        // B 32x16 f16 layout: K = (g>=4?16:0) + laneHalf*8 + (g&3)*2 (pair), N = lane&15
        int kbb = ((g & 4) << 2) + (lg << 3) + ((g & 3) << 1);
        bu[g] = *reinterpret_cast<const uint32_t*>(brow + kbb);
      }
      v16h a = __builtin_bit_cast(v16h, au);
      v16h b = __builtin_bit_cast(v16h, bu);
      acc = __builtin_amdgcn_wmma_f32_16x16x32_f16(
                false, a, false, b, (short)0, acc, false, false);
    }
  }
  __syncthreads();

  // C/D f32 16x16 layout: VGPR g -> M = g + laneHalf*8, N = lane&15
  float* pd = part + (size_t)(r * NSEG + seg) * (T_WIN * NSP);
  #pragma unroll
  for (int g = 0; g < 8; ++g) {
    int t = m0 + mt * 16 + g + (lg << 3);
    int s = nt * 16 + ml;
    if (t < T_WIN) pd[t * NSP + s] = acc[g];
  }
}

// ---------------- Stage 2b: fixed-order reduction of K-split partials ----------------
__global__ void reduce_kernel(const float* __restrict__ part, float* __restrict__ D) {
  int i = blockIdx.x * blockDim.x + threadIdx.x;
  if (i < T_WIN * NSP) {
    float s = 0.0f;
    for (int p = 0; p < NPART; ++p) s += part[(size_t)p * (T_WIN * NSP) + i];
    D[i] = s;   // exact integer regardless of split
  }
}

// ---------------- Stage 3: bin, Fano, MSE (single block, deterministic) ----------------
__global__ __launch_bounds__(256) void fano_mse_kernel(const float* __restrict__ D,
                                                       const float* __restrict__ exp_mean,
                                                       float* __restrict__ out) {
  __shared__ float fan[NBINS][NS];
  __shared__ float sq[NBINS];

  for (int task = threadIdx.x; task < NBINS * NS; task += 256) {
    int b = task / NS, s = task - b * NS;
    int bs = c_bin[b], nb = c_nb[b];
    float sum = 0.0f;
    for (int i = 0; i < nb; ++i) {
      float c = 0.0f;
      int t0 = i * bs;
      for (int u = 0; u < bs; ++u) c += D[(t0 + u) * NSP + s];
      sum += c;
    }
    float mean = sum / (float)nb;
    float vs = 0.0f;
    for (int i = 0; i < nb; ++i) {
      float c = 0.0f;
      int t0 = i * bs;
      for (int u = 0; u < bs; ++u) c += D[(t0 + u) * NSP + s];
      float d = c - mean;
      vs += d * d;
    }
    float var = vs / (float)nb;             // jnp.var: mean of squared deviations
    float den = mean > 1e-7f ? mean : 1e-7f;
    fan[b][s] = var / den;
  }
  __syncthreads();

  if (threadIdx.x < NBINS) {
    int b = threadIdx.x;
    float a = 0.0f;
    for (int s = 0; s < NS; ++s) a += fan[b][s];
    float fm = a / (float)NS;
    float d = exp_mean[b] - fm;
    sq[b] = d * d;
  }
  __syncthreads();

  if (threadIdx.x == 0) {
    float mse = 0.0f;
    for (int b = 0; b < NBINS; ++b) mse += sq[b];
    mse /= (float)NBINS;
    out[0] = 10.0f * mse;    // SYNC_COST * mse
  }
}

extern "C" void kernel_launch(void* const* d_in, const int* in_sizes, int n_in,
                              void* d_out, int out_size, void* d_ws, size_t ws_size,
                              hipStream_t stream) {
  (void)in_sizes; (void)n_in; (void)out_size; (void)ws_size;
  const float*     spikes   = (const float*)d_in[0];      // (8, 600, 30000) f32
  const float*     exp_mean = (const float*)d_in[1];      // (16,) f32
  const int*       trials   = (const int*)d_in[2];        // (50,) i32
  const long long* ids      = (const long long*)d_in[3];  // (50, 200) i64
  const float*     mask     = (const float*)d_in[4];      // (50, 200) f32
  float*           out      = (float*)d_out;              // scalar f32

  // workspace: W f16 (30.72 MB) | partials f32 (48 x 128 KB = 6.14 MB) | D f32 (128 KB)
  _Float16* W = (_Float16*)d_ws;
  const size_t wbytes = (size_t)N_TRIALS * NSP * N_NEUR * sizeof(_Float16);
  float* part = (float*)((char*)d_ws + wbytes);
  float* D    = part + (size_t)NPART * (T_WIN * NSP);

  zero_ws_kernel<<<512, 256, 0, stream>>>((uint32_t*)d_ws, (int)(wbytes / 4));
  build_w_kernel<<<1, 64, 0, stream>>>(trials, ids, mask, W);
  gemm_sel_kernel<<<dim3(16, N_TRIALS, NSEG), 256, 0, stream>>>(spikes, W, part);
  reduce_kernel<<<(T_WIN * NSP + 255) / 256, 256, 0, stream>>>(part, D);
  fano_mse_kernel<<<1, 256, 0, stream>>>(D, exp_mean, out);
}
